// ModeloBaseComprimido_7567732376339
// MI455X (gfx1250) — matrix operand order repaired and checked by reading, not compiled, not run on hardware
//
#include <hip/hip_runtime.h>
#include <hip/hip_bf16.h>

// ---------------------------------------------------------------------------
// Types for CDNA5 WMMA (wave32): D = A(16xK) * B(Kx16) + C, f16 in / f32 acc
// ---------------------------------------------------------------------------
typedef __attribute__((ext_vector_type(16))) _Float16 v16h;
typedef __attribute__((ext_vector_type(8)))  float    v8f;

union V16 {                 // 32-byte fragment per lane (2 x b128 loads)
    uint4    q[2];
    v16h     v;
    _Float16 h[16];
};

#define WMMA_F16(A, B, C) \
    __builtin_amdgcn_wmma_f32_16x16x32_f16(false, (A), false, (B), (short)0, (C), false, false)

// A-fragment (16x32, f16, row-major source): lane L holds row M=L&15.
// halves[0..7]  = K = k0 + 8*(L>=16) + h
// halves[8..15] = K = k0 + 16 + 8*(L>=16) + (h-8)
// => two contiguous 16-byte segments at (k0 + 8*hi) and (k0 + 16 + 8*hi).
__device__ inline v16h load_a_frag_f16(const _Float16* rowbase, int k0, int hi) {
    const _Float16* p = rowbase + k0 + 8 * hi;
    V16 u;
    u.q[0] = *(const uint4*)(p);
    u.q[1] = *(const uint4*)(p + 16);
    return u.v;
}

// B-fragment from packed weights: lane's 16 halves are contiguous.
__device__ inline v16h load_b_frag(const _Float16* fragbase, int lane) {
    const _Float16* p = fragbase + lane * 16;
    V16 u;
    u.q[0] = *(const uint4*)(p);
    u.q[1] = *(const uint4*)(p + 8);
    return u.v;
}

// ---------------------------------------------------------------------------
// Weight packing: f32 KxN row-major  ->  f16 B-fragment-major.
// Fragment (kt, nt), id = nt*(K/32) + kt, 512 halves each.
// Lane L, half h:  W[kt*32 + 8*(L>=16) + (h&7) + 16*(h>=8)][nt*16 + (L&15)]
// (B is Kx16: lane = column N, VGPR v holds K = base+2v, base+2v+1.)
// ---------------------------------------------------------------------------
__global__ void pack_b_frags(const float* __restrict__ W, _Float16* __restrict__ out,
                             int K, int N) {
    int tid = blockIdx.x * blockDim.x + threadIdx.x;
    if (tid >= K * N) return;
    int h    = tid & 15;
    int lane = (tid >> 4) & 31;
    int frag = tid >> 9;
    int kfr  = K >> 5;
    int kt   = frag % kfr;
    int nt   = frag / kfr;
    int k    = kt * 32 + ((lane & 16) ? 8 : 0) + (h & 7) + ((h & 8) ? 16 : 0);
    int n    = nt * 16 + (lane & 15);
    out[tid] = (_Float16)W[(size_t)k * N + n];
}

__global__ void init_state(const float* __restrict__ src, float* __restrict__ dstf,
                           _Float16* __restrict__ dsth, int n) {
    int i = blockIdx.x * blockDim.x + threadIdx.x;
    if (i < n) { float v = src[i]; dstf[i] = v; dsth[i] = (_Float16)v; }
}

__global__ void zero_f32(float* __restrict__ p, int n) {
    int i = blockIdx.x * blockDim.x + threadIdx.x;
    if (i < n) p[i] = 0.0f;
}

// ---------------------------------------------------------------------------
// Message kernel: one wave handles 32 message rows (2 A row-tiles) so every
// packed B fragment feeds TWO wmma ops (register-level B reuse, halves the
// VMEM-issue : WMMA ratio and the L2 weight traffic).
//   hidden = relu([ls[first] | ls[second]] @ W1 + b1)   (32x256, 256 WMMAs)
//   msg    = relu(hidden @ W2 + b2)                     (32x128, 128 WMMAs)
//   edges[second[m]] += msg[m]   (f32 global atomics, segment_sum)
// ---------------------------------------------------------------------------
__global__ void __launch_bounds__(32) message_kernel(
    const _Float16* __restrict__ ls_h,        // [E,128] f16 state
    const int* __restrict__ first, const int* __restrict__ second,
    const _Float16* __restrict__ W1p,         // packed, K=256 N=256 (kfr=8)
    const float* __restrict__ b1,             // [256]
    const _Float16* __restrict__ W2p,         // packed, K=256 N=128 (kfr=8)
    const float* __restrict__ b2,             // [128]
    float* __restrict__ edges,                // [E,128] accumulator
    int M)
{
    __shared__ _Float16 sh[32 * 256];         // hidden, row-major f16 (16 KB)
    __shared__ int sdst[32];

    int lane = threadIdx.x;                   // 0..31
    int m0   = blockIdx.x * 32;
    if (m0 >= M) return;
    int r  = lane & 15;
    int hi = lane >> 4;

    int mA = m0 + r;      if (mA >= M) mA = M - 1;    // row-set 0: rows 0..15
    int mB = m0 + 16 + r; if (mB >= M) mB = M - 1;    // row-set 1: rows 16..31
    sdst[lane] = second[(m0 + lane < M) ? (m0 + lane) : (M - 1)];

    const _Float16* r1a = ls_h + (size_t)first[mA]  * 128;
    const _Float16* r2a = ls_h + (size_t)second[mA] * 128;
    const _Float16* r1b = ls_h + (size_t)first[mB]  * 128;
    const _Float16* r2b = ls_h + (size_t)second[mB] * 128;

    // A fragments over the 256-wide concat: cols 0..127 -> first, 128..255 -> second
    v16h a0[8], a1[8];
#pragma unroll
    for (int kc = 0; kc < 8; ++kc) {
        a0[kc] = (kc < 4) ? load_a_frag_f16(r1a, kc * 32, hi)
                          : load_a_frag_f16(r2a, (kc - 4) * 32, hi);
        a1[kc] = (kc < 4) ? load_a_frag_f16(r1b, kc * 32, hi)
                          : load_a_frag_f16(r2b, (kc - 4) * 32, hi);
    }

    // ---- layer 1: hidden[32,256] ----
    for (int nt = 0; nt < 16; ++nt) {
        float bv = b1[nt * 16 + r];
        v8f acc0 = {bv, bv, bv, bv, bv, bv, bv, bv};
        v8f acc1 = acc0;
        const _Float16* wb = W1p + (size_t)(nt * 8) * 512;
#pragma unroll
        for (int kc = 0; kc < 8; ++kc) {
            v16h b = load_b_frag(wb + (size_t)kc * 512, lane);   // loaded once,
            acc0 = WMMA_F16(a0[kc], b, acc0);                    // used twice
            acc1 = WMMA_F16(a1[kc], b, acc1);
        }
        // relu + transpose through LDS: lane holds col n=nt*16+r, rows v+8*hi
        int n = nt * 16 + r;
#pragma unroll
        for (int v = 0; v < 8; ++v) {
            float x0 = acc0[v] > 0.0f ? acc0[v] : 0.0f;
            float x1 = acc1[v] > 0.0f ? acc1[v] : 0.0f;
            sh[(v + 8 * hi) * 256 + n]        = (_Float16)x0;
            sh[(16 + v + 8 * hi) * 256 + n]   = (_Float16)x1;
        }
    }
    __syncthreads();

    // preload layer-2 A fragments from LDS once (b128 ds loads)
    v16h h0[8], h1[8];
#pragma unroll
    for (int kc = 0; kc < 8; ++kc) {
        h0[kc] = load_a_frag_f16(sh + r * 256,        kc * 32, hi);
        h1[kc] = load_a_frag_f16(sh + (16 + r) * 256, kc * 32, hi);
    }

    // ---- layer 2 + scatter: msg[32,128] ----
    for (int nt = 0; nt < 8; ++nt) {
        float bv = b2[nt * 16 + r];
        v8f acc0 = {bv, bv, bv, bv, bv, bv, bv, bv};
        v8f acc1 = acc0;
        const _Float16* wb = W2p + (size_t)(nt * 8) * 512;
#pragma unroll
        for (int kc = 0; kc < 8; ++kc) {
            v16h b = load_b_frag(wb + (size_t)kc * 512, lane);
            acc0 = WMMA_F16(h0[kc], b, acc0);
            acc1 = WMMA_F16(h1[kc], b, acc1);
        }
        int n = nt * 16 + r;
#pragma unroll
        for (int v = 0; v < 8; ++v) {
            float x0 = acc0[v] > 0.0f ? acc0[v] : 0.0f;
            float x1 = acc1[v] > 0.0f ? acc1[v] : 0.0f;
            int row0 = v + 8 * hi;
            int row1 = 16 + v + 8 * hi;
            if (m0 + row0 < M)
                atomicAdd(&edges[(size_t)sdst[row0] * 128 + n], x0);
            if (m0 + row1 < M)
                atomicAdd(&edges[(size_t)sdst[row1] * 128 + n], x1);
        }
    }
}

// ---------------------------------------------------------------------------
// GRU kernel: one wave handles 16 edge rows (in-place update; wave owns rows).
//   mx = x @ kernel + b[0];  mh = h @ recurrent + b[1]   (192 WMMAs)
//   z = sig(xz+hz); r = sig(xr+hr); hc = tanh(xh + r*hh)
//   h' = z*h + (1-z)*hc
// (6 live f32 accumulators + 8 A frags; 32-row tiling here would spill.)
// ---------------------------------------------------------------------------
__global__ void __launch_bounds__(32) gru_kernel(
    const float* __restrict__ edges,          // [E,128] x input (f32)
    float* __restrict__ ls_f,                 // [E,128] h (in/out, f32)
    _Float16* __restrict__ ls_h,              // [E,128] h (in/out, f16)
    const _Float16* __restrict__ GKp,         // packed, K=128 N=384 (kfr=4)
    const _Float16* __restrict__ GRp,         // packed, K=128 N=384 (kfr=4)
    const float* __restrict__ gbias,          // [2,384], gate order z,r,h
    int E)
{
    int lane = threadIdx.x;
    int e0   = blockIdx.x * 16;
    if (e0 >= E) return;
    int r  = lane & 15;
    int hi = lane >> 4;

    const float*    xrow = edges + (size_t)(e0 + r) * 128;
    const _Float16* hrow = ls_h  + (size_t)(e0 + r) * 128;

    v16h xa[4], ha[4];
#pragma unroll
    for (int kc = 0; kc < 4; ++kc) {
        const float* p = xrow + kc * 32 + 8 * hi;
        V16 u;
#pragma unroll
        for (int t = 0; t < 8; ++t) {
            u.h[t]     = (_Float16)p[t];
            u.h[8 + t] = (_Float16)p[16 + t];
        }
        xa[kc] = u.v;
        ha[kc] = load_a_frag_f16(hrow, kc * 32, hi);
    }

    for (int j = 0; j < 8; ++j) {             // tiles over D=128 columns
        int col = j * 16 + r;
        float b0z = gbias[col],       b0r = gbias[128 + col], b0h = gbias[256 + col];
        float b1z = gbias[384 + col], b1r = gbias[512 + col], b1h = gbias[640 + col];
        v8f xz = {b0z,b0z,b0z,b0z,b0z,b0z,b0z,b0z};
        v8f xr = {b0r,b0r,b0r,b0r,b0r,b0r,b0r,b0r};
        v8f xh = {b0h,b0h,b0h,b0h,b0h,b0h,b0h,b0h};
        v8f hz = {b1z,b1z,b1z,b1z,b1z,b1z,b1z,b1z};
        v8f hr = {b1r,b1r,b1r,b1r,b1r,b1r,b1r,b1r};
        v8f hh = {b1h,b1h,b1h,b1h,b1h,b1h,b1h,b1h};
#pragma unroll
        for (int kc = 0; kc < 4; ++kc) {
            v16h bz = load_b_frag(GKp + (size_t)((j)      * 4 + kc) * 512, lane);
            v16h br = load_b_frag(GKp + (size_t)((j + 8)  * 4 + kc) * 512, lane);
            v16h bh = load_b_frag(GKp + (size_t)((j + 16) * 4 + kc) * 512, lane);
            xz = WMMA_F16(xa[kc], bz, xz);
            xr = WMMA_F16(xa[kc], br, xr);
            xh = WMMA_F16(xa[kc], bh, xh);
            v16h cz = load_b_frag(GRp + (size_t)((j)      * 4 + kc) * 512, lane);
            v16h cr = load_b_frag(GRp + (size_t)((j + 8)  * 4 + kc) * 512, lane);
            v16h ch = load_b_frag(GRp + (size_t)((j + 16) * 4 + kc) * 512, lane);
            hz = WMMA_F16(ha[kc], cz, hz);
            hr = WMMA_F16(ha[kc], cr, hr);
            hh = WMMA_F16(ha[kc], ch, hh);
        }
#pragma unroll
        for (int v = 0; v < 8; ++v) {
            int row = e0 + v + 8 * hi;
            size_t idx = (size_t)row * 128 + col;
            float z  = 1.0f / (1.0f + __expf(-(xz[v] + hz[v])));
            float rr = 1.0f / (1.0f + __expf(-(xr[v] + hr[v])));
            float hc = tanhf(xh[v] + rr * hh[v]);
            float hold = ls_f[idx];
            float hnew = z * hold + (1.0f - z) * hc;
            ls_f[idx] = hnew;
            ls_h[idx] = (_Float16)hnew;
        }
    }
}

// ---------------------------------------------------------------------------
// Graph segment-sum and tiny readout MLP (negligible FLOPs -> scalar kernels)
// ---------------------------------------------------------------------------
__global__ void graph_sum(const float* __restrict__ ls_f, const int* __restrict__ gid,
                          float* __restrict__ gs, int E) {
    int i = blockIdx.x * blockDim.x + threadIdx.x;
    if (i >= E * 128) return;
    int e = i >> 7, c = i & 127;
    atomicAdd(&gs[(size_t)gid[e] * 128 + c], ls_f[i]);
}

__global__ void dense_kernel(const float* __restrict__ A, const float* __restrict__ W,
                             const float* __restrict__ b, float* __restrict__ out,
                             int rows, int K, int N, int do_relu) {
    int i = blockIdx.x * blockDim.x + threadIdx.x;
    if (i >= rows * N) return;
    int rr = i / N, n = i % N;
    float acc = b[n];
    for (int k = 0; k < K; ++k) acc += A[(size_t)rr * K + k] * W[(size_t)k * N + n];
    out[i] = do_relu ? fmaxf(acc, 0.0f) : acc;
}

// ---------------------------------------------------------------------------
// Host launcher
// ---------------------------------------------------------------------------
extern "C" void kernel_launch(void* const* d_in, const int* in_sizes, int n_in,
                              void* d_out, int out_size, void* d_ws, size_t ws_size,
                              hipStream_t stream) {
    const float* link_state = (const float*)d_in[0];
    const int*   first      = (const int*)d_in[1];
    const int*   second     = (const int*)d_in[2];
    const int*   gids       = (const int*)d_in[3];
    const float* mW1 = (const float*)d_in[5];
    const float* mb1 = (const float*)d_in[6];
    const float* mW2 = (const float*)d_in[7];
    const float* mb2 = (const float*)d_in[8];
    const float* gk  = (const float*)d_in[9];
    const float* gr  = (const float*)d_in[10];
    const float* gb  = (const float*)d_in[11];
    const float* rW1 = (const float*)d_in[12];
    const float* rb1 = (const float*)d_in[13];
    const float* rW2 = (const float*)d_in[14];
    const float* rb2 = (const float*)d_in[15];
    const float* rW3 = (const float*)d_in[16];
    const float* rb3 = (const float*)d_in[17];

    const int E = in_sizes[0] / 128;
    const int M = in_sizes[1];
    const int T = 3;
    const int G = 64;

    char*  ws  = (char*)d_ws;
    size_t off = 0;
    auto alloc = [&](size_t bytes) -> void* {
        void* p = ws + off;
        off += (bytes + 255) & ~(size_t)255;
        return p;
    };
    float*    ls_f  = (float*)   alloc((size_t)E * 128 * 4);
    _Float16* ls_h  = (_Float16*)alloc((size_t)E * 128 * 2);
    float*    edges = (float*)   alloc((size_t)E * 128 * 4);
    _Float16* W1p   = (_Float16*)alloc((size_t)256 * 256 * 2);
    _Float16* W2p   = (_Float16*)alloc((size_t)256 * 128 * 2);
    _Float16* GKp   = (_Float16*)alloc((size_t)128 * 384 * 2);
    _Float16* GRp   = (_Float16*)alloc((size_t)128 * 384 * 2);
    float*    gs    = (float*)   alloc((size_t)G * 128 * 4);
    float*    r1    = (float*)   alloc((size_t)G * 256 * 4);
    float*    r2    = (float*)   alloc((size_t)G * 256 * 4);

    int n = E * 128;
    init_state<<<(n + 255) / 256, 256, 0, stream>>>(link_state, ls_f, ls_h, n);
    pack_b_frags<<<(256 * 256 + 255) / 256, 256, 0, stream>>>(mW1, W1p, 256, 256);
    pack_b_frags<<<(256 * 128 + 255) / 256, 256, 0, stream>>>(mW2, W2p, 256, 128);
    pack_b_frags<<<(128 * 384 + 255) / 256, 256, 0, stream>>>(gk,  GKp, 128, 384);
    pack_b_frags<<<(128 * 384 + 255) / 256, 256, 0, stream>>>(gr,  GRp, 128, 384);

    for (int t = 0; t < T; ++t) {
        zero_f32<<<(n + 255) / 256, 256, 0, stream>>>(edges, n);
        message_kernel<<<(M + 31) / 32, 32, 0, stream>>>(
            ls_h, first, second, W1p, mb1, W2p, mb2, edges, M);
        gru_kernel<<<(E + 15) / 16, 32, 0, stream>>>(
            edges, ls_f, ls_h, GKp, GRp, gb, E);
    }

    zero_f32<<<(G * 128 + 255) / 256, 256, 0, stream>>>(gs, G * 128);
    graph_sum<<<(n + 255) / 256, 256, 0, stream>>>(ls_f, gids, gs, E);

    dense_kernel<<<(G * 256 + 255) / 256, 256, 0, stream>>>(gs, rW1, rb1, r1, G, 128, 256, 1);
    dense_kernel<<<(G * 256 + 255) / 256, 256, 0, stream>>>(r1, rW2, rb2, r2, G, 256, 256, 1);
    dense_kernel<<<(G + 255) / 256, 256, 0, stream>>>(r2, rW3, rb3, (float*)d_out, G, 256, 1, 0);
}
